// NeighbourEmbedding_14474039787543
// MI455X (gfx1250) — matrix operand order repaired and checked by reading, not compiled, not run on hardware
//
#include <hip/hip_runtime.h>

// ---------------------------------------------------------------------------
// Types
// ---------------------------------------------------------------------------
typedef __bf16 bf16_t;
typedef __bf16 v8bf  __attribute__((ext_vector_type(8)));
typedef __bf16 v16bf __attribute__((ext_vector_type(16)));
typedef float  v8f   __attribute__((ext_vector_type(8)));
typedef unsigned int v4u __attribute__((ext_vector_type(4)));
typedef int v8i __attribute__((ext_vector_type(8)));
typedef int v4i __attribute__((ext_vector_type(4)));

union ABf { v16bf v; v8bf h[2]; };

// ---------------------------------------------------------------------------
// TDM: build D# per CDNA5 ISA and issue tensor_load_to_lds
// 2D tile: tile_dim0 elements (contiguous), tile_dim1 rows, row stride stride0.
// data_size = 2 bytes (bf16).  This toolchain's builtin takes 6 operands
// (g0 4dw, g1 8dw, g2 4dw, g3 4dw, extra 8dw, cpol); our tensor is 2-D so the
// trailing groups are zero (tile_dim2/3 = 0 -> unused per ISA).
// ---------------------------------------------------------------------------
__device__ inline void tdm_load_2d_bf16(unsigned lds_off, const void* gptr,
                                        unsigned tile0, unsigned tile1,
                                        unsigned long long stride0,
                                        unsigned tdim0, unsigned tdim1)
{
  unsigned long long ga = (unsigned long long)(size_t)gptr;
  v4u g0;
  g0[0] = 1u;                                   // count=1, user descriptor
  g0[1] = lds_off;                              // lds_addr
  g0[2] = (unsigned)(ga & 0xFFFFFFFFu);         // global_addr[31:0]
  g0[3] = ((unsigned)((ga >> 32) & 0x1FFFFFFu)) | 0x80000000u; // [56:32], type=2
  v8i g1;
  g1[0] = (int)(1u << 16);                      // workgroup_mask=0, data_size=1 (2B)
  g1[1] = (int)((tdim0 & 0xFFFFu) << 16);       // tensor_dim0[15:0]
  g1[2] = (int)(((tdim0 >> 16) & 0xFFFFu) | ((tdim1 & 0xFFFFu) << 16));
  g1[3] = (int)(((tdim1 >> 16) & 0xFFFFu) | ((tile0 & 0xFFFFu) << 16));
  g1[4] = (int)(tile1 & 0xFFFFu);               // tile_dim1, tile_dim2=0
  g1[5] = (int)(stride0 & 0xFFFFFFFFull);       // tensor_dim0_stride[31:0]
  g1[6] = (int)((stride0 >> 32) & 0xFFFFull);   // stride0[47:32], stride1=0
  g1[7] = 0;
  v4i g2 = {0, 0, 0, 0};
  v4i g3 = {0, 0, 0, 0};
  v8i g4 = {0, 0, 0, 0, 0, 0, 0, 0};
  __builtin_amdgcn_tensor_load_to_lds(g0, g1, g2, g3, g4, 0);
}

// ---------------------------------------------------------------------------
// WMMA GEMM:  Y[Nt][M] = X[Nt][K] * W[M][K]^T   (bf16 in, f32 acc, bf16 out)
// Block: 256 threads (8 waves), 128 points per block; W + X block staged in LDS
// via TDM.  Per CDNA5 ISA layouts, every fragment load is a
// contiguous 16B LDS read.
// ---------------------------------------------------------------------------
template<int M, int K>
__global__ __launch_bounds__(256) void gemm_wmma_kernel(
    const bf16_t* __restrict__ X, const bf16_t* __restrict__ Wm,
    bf16_t* __restrict__ Y, long long Nt)
{
  extern __shared__ char smem_raw[];
  bf16_t* sX = (bf16_t*)smem_raw;     // [128][K]
  bf16_t* sW = sX + 128 * K;          // [M][K]
  const long long p0 = (long long)blockIdx.x * 128;

  if (threadIdx.x < 32) {             // wave 0 issues both DMAs
    unsigned ldsX = (unsigned)(size_t)(void*)sX;
    unsigned ldsW = (unsigned)(size_t)(void*)sW;
    tdm_load_2d_bf16(ldsX, X + p0 * K, K, 128, K, K, 128);
    tdm_load_2d_bf16(ldsW, Wm,         K, M,   K, K, M);
    __builtin_amdgcn_s_wait_tensorcnt(0);
  }
  __syncthreads();

  const int wave = threadIdx.x >> 5;  // point-tile id (0..7)
  const int lane = threadIdx.x & 31;
  const int lh   = lane >> 4;         // half-wave select
  const int lm   = lane & 15;

  // A fragments (activations) for this wave's 16 points, all K chunks.
  ABf a[K / 32];
  const bf16_t* xrow = sX + (wave * 16 + lm) * K;
#pragma unroll
  for (int kc = 0; kc < K / 32; ++kc) {
    const int kb = kc * 32 + lh * 8;        // lanes<16: K 0..7 / 16..23; >=16: +8
    a[kc].h[0] = *(const v8bf*)(xrow + kb);
    a[kc].h[1] = *(const v8bf*)(xrow + kb + 16);
  }

#pragma unroll
  for (int mt = 0; mt < M / 16; ++mt) {
    v8f acc = {0.f, 0.f, 0.f, 0.f, 0.f, 0.f, 0.f, 0.f};
    const bf16_t* wrow = sW + (mt * 16 + lm) * K;
#pragma unroll
    for (int kc = 0; kc < K / 32; ++kc) {
      ABf bb;
      const int kb = kc * 32 + lh * 8;      // lanes<16: K 0..7 / 16..23; >=16: +8
      bb.h[0] = *(const v8bf*)(wrow + kb);
      bb.h[1] = *(const v8bf*)(wrow + kb + 16);
      acc = __builtin_amdgcn_wmma_f32_16x16x32_bf16(
          false, a[kc].v, false, bb.v, (short)0, acc, false, false);
    }
    const long long prow = p0 + wave * 16 + lh * 8;
#pragma unroll
    for (int r = 0; r < 8; ++r)
      Y[(prow + r) * M + mt * 16 + lm] = (bf16_t)acc[r];
  }
}

// ---------------------------------------------------------------------------
// Elementwise / reduction helpers
// ---------------------------------------------------------------------------
__global__ __launch_bounds__(256) void coords_kernel(
    const float* __restrict__ x, float* __restrict__ coords, int N, long long total)
{
  long long p = (long long)blockIdx.x * 256 + threadIdx.x;
  if (p >= total) return;
  long long b = p / N, l = p % N;
  const float* xb = x + (b * 3) * (long long)N + l;
  coords[p * 3 + 0] = xb[0];
  coords[p * 3 + 1] = xb[(long long)N];
  coords[p * 3 + 2] = xb[2LL * N];
}

__global__ __launch_bounds__(256) void conv1_kernel(
    const float* __restrict__ x, const float* __restrict__ w1,
    bf16_t* __restrict__ F0, int N, long long total)
{
  __shared__ float ws1[192];
  int t = threadIdx.x;
  if (t < 192) ws1[t] = w1[t];
  __syncthreads();
  int c = t & 63;
  long long p = (long long)blockIdx.x * 4 + (t >> 6);
  if (p >= total) return;
  long long b = p / N, l = p % N;
  const float* xb = x + (b * 3) * (long long)N + l;
  float v = ws1[c * 3] * xb[0] + ws1[c * 3 + 1] * xb[(long long)N]
          + ws1[c * 3 + 2] * xb[2LL * N];
  F0[p * 64 + c] = (bf16_t)v;
}

__global__ __launch_bounds__(256) void castw_kernel(
    const float* __restrict__ w, bf16_t* __restrict__ wb, int n)
{
  int i = blockIdx.x * 256 + threadIdx.x;
  if (i < n) wb[i] = (bf16_t)w[i];
}

// Deterministic per-block partial channel sums/sumsqs (no float atomics).
__global__ __launch_bounds__(256) void stats_partial_kernel(
    const bf16_t* __restrict__ Y, long long Nt, int M, int rowsPerBlock,
    float* __restrict__ part)
{
  __shared__ float ls[256], lq[256];
  int t = threadIdx.x;
  int c = t % M;
  int rsub = t / M;
  int rpi = 256 / M;
  long long r0 = (long long)blockIdx.x * rowsPerBlock;
  long long r1 = r0 + rowsPerBlock; if (r1 > Nt) r1 = Nt;
  float s = 0.f, q = 0.f;
  for (long long r = r0 + rsub; r < r1; r += rpi) {
    float v = (float)Y[r * M + c];
    s += v; q += v * v;
  }
  ls[t] = s; lq[t] = q;
  __syncthreads();
  if (t < M) {
    float ss = 0.f, qq = 0.f;
    for (int j = 0; j < rpi; ++j) { ss += ls[t + j * M]; qq += lq[t + j * M]; }
    part[(long long)blockIdx.x * (2 * M) + t]     = ss;
    part[(long long)blockIdx.x * (2 * M) + M + t] = qq;
  }
}

__global__ void stats_final_kernel(const float* __restrict__ part, int nblk,
                                   int M, float* __restrict__ stats)
{
  int j = blockIdx.x * blockDim.x + threadIdx.x;
  if (j < 2 * M) {
    float s = 0.f;
    for (int b = 0; b < nblk; ++b) s += part[(long long)b * (2 * M) + j];
    stats[j] = s;
  }
}

__global__ __launch_bounds__(256) void bnrelu_kernel(
    bf16_t* __restrict__ Y, const float* __restrict__ stats,
    const float* __restrict__ gamma, const float* __restrict__ beta,
    long long total, int M, float invCnt)
{
  long long i = (long long)blockIdx.x * 256 + threadIdx.x;
  if (i >= total) return;
  int c = (int)(i % M);
  float mean = stats[c] * invCnt;
  float var  = stats[M + c] * invCnt - mean * mean;
  float rstd = rsqrtf(var + 1e-5f);
  float v = ((float)Y[i] - mean) * rstd * gamma[c] + beta[c];
  Y[i] = (bf16_t)fmaxf(v, 0.f);
}

// BN + ReLU + max over k=32 group members. finalMode writes f32 [B][M][S].
__global__ void bnmax_kernel(
    const bf16_t* __restrict__ Y, const float* __restrict__ stats,
    const float* __restrict__ gamma, const float* __restrict__ beta,
    float invCnt, int M, int S, int finalMode,
    bf16_t* __restrict__ outb, float* __restrict__ outf)
{
  int c = threadIdx.x;            // blockDim.x == M
  long long g = blockIdx.x;
  float mean = stats[c] * invCnt;
  float rstd = rsqrtf(stats[M + c] * invCnt - mean * mean + 1e-5f);
  float ga = gamma[c], be = beta[c];
  float m = 0.f;                  // relu output is >= 0
  const bf16_t* Yg = Y + g * 32 * M + c;
#pragma unroll
  for (int i = 0; i < 32; ++i) {
    float v = ((float)Yg[(long long)i * M] - mean) * rstd * ga + be;
    m = fmaxf(m, fmaxf(v, 0.f));
  }
  if (finalMode) {
    long long b = g / S, s = g % S;
    outf[(b * M + c) * (long long)S + s] = m;
  } else {
    outb[g * M + c] = (bf16_t)m;
  }
}

// ---------------------------------------------------------------------------
// FPS: one workgroup per batch; coords + running min-dist resident in LDS.
// 511 sequential argmax iterations (matches jnp scan, start index 0).
// ---------------------------------------------------------------------------
__global__ __launch_bounds__(256) void fps_kernel(
    const float* __restrict__ coords, int N, int S,
    int* __restrict__ idxOut, float* __restrict__ xyzOut)
{
  extern __shared__ char smem_raw[];
  float* cx = (float*)smem_raw;
  float* cy = cx + N; float* cz = cy + N; float* dist = cz + N;
  __shared__ float rd[256];
  __shared__ int   ri[256];
  __shared__ int   slast;
  int b = blockIdx.x, t = threadIdx.x;
  const float* cb = coords + (long long)b * N * 3;
  for (int n = t; n < N; n += 256) {
    cx[n] = cb[3 * n]; cy[n] = cb[3 * n + 1]; cz[n] = cb[3 * n + 2];
    dist[n] = 1e10f;
  }
  if (t == 0) {
    slast = 0;
    idxOut[(long long)b * S] = 0;
    xyzOut[((long long)b * S) * 3 + 0] = cb[0];
    xyzOut[((long long)b * S) * 3 + 1] = cb[1];
    xyzOut[((long long)b * S) * 3 + 2] = cb[2];
  }
  __syncthreads();
  for (int it = 1; it < S; ++it) {
    int last = slast;
    float bx = cx[last], by = cy[last], bz = cz[last];
    float bd = -1.f; int bi = 0;
    for (int n = t; n < N; n += 256) {
      float dx = cx[n] - bx, dy = cy[n] - by, dz = cz[n] - bz;
      float d = dx * dx + dy * dy + dz * dz;
      float dn = fminf(dist[n], d);
      dist[n] = dn;
      if (dn > bd) { bd = dn; bi = n; }
    }
    rd[t] = bd; ri[t] = bi;
    __syncthreads();
    for (int o = 128; o > 0; o >>= 1) {
      if (t < o) {
        float d2 = rd[t + o]; int i2 = ri[t + o];
        if (d2 > rd[t] || (d2 == rd[t] && i2 < ri[t])) { rd[t] = d2; ri[t] = i2; }
      }
      __syncthreads();
    }
    if (t == 0) {
      int w = ri[0];
      slast = w;
      idxOut[(long long)b * S + it] = w;
      xyzOut[((long long)b * S + it) * 3 + 0] = cx[w];
      xyzOut[((long long)b * S + it) * 3 + 1] = cy[w];
      xyzOut[((long long)b * S + it) * 3 + 2] = cz[w];
    }
    __syncthreads();
  }
}

// ---------------------------------------------------------------------------
// kNN (k=32): one thread per query, batch coords in LDS, fully-unrolled
// register insertion list (compile-time indices -> stays in VGPRs).
// ---------------------------------------------------------------------------
__global__ __launch_bounds__(256) void knn_kernel(
    const float* __restrict__ coords, const float* __restrict__ q,
    int N, int S, int* __restrict__ knnOut)
{
  extern __shared__ char smem_raw[];
  float* cx = (float*)smem_raw;
  float* cy = cx + N; float* cz = cy + N;
  int b = blockIdx.y;
  const float* cb = coords + (long long)b * N * 3;
  for (int n = threadIdx.x; n < N; n += 256) {
    cx[n] = cb[3 * n]; cy[n] = cb[3 * n + 1]; cz[n] = cb[3 * n + 2];
  }
  __syncthreads();
  int s = blockIdx.x * 256 + threadIdx.x;
  long long qi = (long long)b * S + s;
  float qx = q[qi * 3], qy = q[qi * 3 + 1], qz = q[qi * 3 + 2];
  float dl[32]; int il[32];
#pragma unroll
  for (int j = 0; j < 32; ++j) { dl[j] = 3.0e38f; il[j] = 0; }
  for (int n = 0; n < N; ++n) {
    float dx = cx[n] - qx, dy = cy[n] - qy, dz = cz[n] - qz;
    float d = dx * dx + dy * dy + dz * dz;
    if (d < dl[31]) {
#pragma unroll
      for (int j = 31; j > 0; --j) {
        bool mv  = d < dl[j - 1];
        float nd = mv ? dl[j - 1] : d;
        int   ni = mv ? il[j - 1] : n;
        bool upd = d < dl[j];
        dl[j] = upd ? nd : dl[j];
        il[j] = upd ? ni : il[j];
      }
      if (d < dl[0]) { dl[0] = d; il[0] = n; }
    }
  }
#pragma unroll
  for (int j = 0; j < 32; ++j) knnOut[qi * 32 + j] = il[j];
}

// ---------------------------------------------------------------------------
// Grouping: row r = (group g, neighbour i); channels [0,D)=neigh-center,
// [D,2D)=center.  Output point-major bf16 ready for the WMMA GEMM.
// ---------------------------------------------------------------------------
__global__ void group_kernel(
    const bf16_t* __restrict__ F, const int* __restrict__ fpsIdx,
    const int* __restrict__ knnIdx, bf16_t* __restrict__ Xg,
    int D, int S, int Npts)
{
  long long r = blockIdx.x;
  int c = threadIdx.x;            // blockDim.x == 2*D
  long long g = r >> 5;
  int i = (int)(r & 31);
  long long b = g / S;
  int nidx = knnIdx[g * 32 + i];
  int cidx = fpsIdx[g];
  const bf16_t* Fn = F + (b * Npts + nidx) * (long long)D;
  const bf16_t* Fc = F + (b * Npts + cidx) * (long long)D;
  float v = (c < D) ? ((float)Fn[c] - (float)Fc[c]) : (float)Fc[c - D];
  Xg[r * (2 * D) + c] = (bf16_t)v;
}

// ---------------------------------------------------------------------------
// Host orchestration
// ---------------------------------------------------------------------------
extern "C" void kernel_launch(void* const* d_in, const int* in_sizes, int n_in,
                              void* d_out, int out_size, void* d_ws, size_t ws_size,
                              hipStream_t stream) {
  (void)in_sizes; (void)n_in; (void)out_size; (void)ws_size;
  const int Bn = 32, N0 = 8192, S1 = 512, S2 = 256;
  const long long BN  = (long long)Bn * N0;       // 262144
  const long long Nt1 = (long long)Bn * S1 * 32;  // 524288
  const long long Nt2 = (long long)Bn * S2 * 32;  // 262144

  const float* x    = (const float*)d_in[0];
  const float* w1   = (const float*)d_in[1];
  const float* g1   = (const float*)d_in[2];
  const float* b1   = (const float*)d_in[3];
  const float* w2   = (const float*)d_in[4];
  const float* g2   = (const float*)d_in[5];
  const float* b2   = (const float*)d_in[6];
  const float* s1wa = (const float*)d_in[7];
  const float* s1ga = (const float*)d_in[8];
  const float* s1ba = (const float*)d_in[9];
  const float* s1wb = (const float*)d_in[10];
  const float* s1gb = (const float*)d_in[11];
  const float* s1bb = (const float*)d_in[12];
  const float* s2wa = (const float*)d_in[13];
  const float* s2ga = (const float*)d_in[14];
  const float* s2ba = (const float*)d_in[15];
  const float* s2wb = (const float*)d_in[16];
  const float* s2gb = (const float*)d_in[17];
  const float* s2bb = (const float*)d_in[18];
  float* out = (float*)d_out;

  char* ws = (char*)d_ws;
  size_t off = 0;
  auto alloc = [&](size_t bytes) {
    size_t o = off; off += (bytes + 255) & ~(size_t)255; return o;
  };
  float* coords = (float*)(ws + alloc((size_t)BN * 3 * 4));
  float* xyz1   = (float*)(ws + alloc((size_t)Bn * S1 * 3 * 4));
  float* xyz2   = (float*)(ws + alloc((size_t)Bn * S2 * 3 * 4));
  int*   fps1   = (int*)  (ws + alloc((size_t)Bn * S1 * 4));
  int*   fps2   = (int*)  (ws + alloc((size_t)Bn * S2 * 4));
  int*   knn1   = (int*)  (ws + alloc((size_t)Bn * S1 * 32 * 4));
  int*   knn2   = (int*)  (ws + alloc((size_t)Bn * S2 * 32 * 4));
  float* stats  = (float*)(ws + alloc(512 * 4));
  float* part   = (float*)(ws + alloc(64 * 512 * 4));
  bf16_t* wbf   = (bf16_t*)(ws + alloc(256 * 256 * 2));
  bf16_t* f1pt  = (bf16_t*)(ws + alloc((size_t)Bn * S1 * 128 * 2));
  bf16_t* bufA  = (bf16_t*)(ws + alloc((size_t)134217728));
  bf16_t* bufB  = (bf16_t*)(ws + alloc((size_t)134217728));

  const int ROWS = 8192;
  auto bn_stage = [&](bf16_t* Y, long long Nt, int M, const float* ga,
                      const float* be) {
    int nblk = (int)((Nt + ROWS - 1) / ROWS);
    stats_partial_kernel<<<nblk, 256, 0, stream>>>(Y, Nt, M, ROWS, part);
    stats_final_kernel<<<1, 512, 0, stream>>>(part, nblk, M, stats);
    long long tot = Nt * M;
    bnrelu_kernel<<<(unsigned)((tot + 255) / 256), 256, 0, stream>>>(
        Y, stats, ga, be, tot, M, 1.0f / (float)Nt);
  };

  // ---- stage 0: coords + conv1 + BN/ReLU ----
  coords_kernel<<<(unsigned)((BN + 255) / 256), 256, 0, stream>>>(x, coords, N0, BN);
  conv1_kernel<<<(unsigned)(BN / 4), 256, 0, stream>>>(x, w1, bufA, N0, BN);
  bn_stage(bufA, BN, 64, g1, b1);

  // ---- conv2 (WMMA GEMM 64x64) + BN/ReLU ----
  castw_kernel<<<(64 * 64 + 255) / 256, 256, 0, stream>>>(w2, wbf, 64 * 64);
  gemm_wmma_kernel<64, 64><<<(unsigned)(BN / 128), 256,
      (128 * 64 + 64 * 64) * 2, stream>>>(bufA, wbf, bufB, BN);
  bn_stage(bufB, BN, 64, g2, b2);

  // ---- SG block 1: FPS / kNN / group ----
  fps_kernel<<<Bn, 256, (size_t)4 * N0 * 4, stream>>>(coords, N0, S1, fps1, xyz1);
  knn_kernel<<<dim3(S1 / 256, Bn), 256, (size_t)3 * N0 * 4, stream>>>(
      coords, xyz1, N0, S1, knn1);
  group_kernel<<<(unsigned)Nt1, 128, 0, stream>>>(bufB, fps1, knn1, bufA, 64, S1, N0);

  // ---- SG1 conv a (128x128) + BN/ReLU ----
  castw_kernel<<<(128 * 128 + 255) / 256, 256, 0, stream>>>(s1wa, wbf, 128 * 128);
  gemm_wmma_kernel<128, 128><<<(unsigned)(Nt1 / 128), 256,
      (128 * 128 + 128 * 128) * 2, stream>>>(bufA, wbf, bufB, Nt1);
  bn_stage(bufB, Nt1, 128, s1ga, s1ba);

  // ---- SG1 conv b (128x128) + BN/ReLU/maxpool -> f1 ----
  castw_kernel<<<(128 * 128 + 255) / 256, 256, 0, stream>>>(s1wb, wbf, 128 * 128);
  gemm_wmma_kernel<128, 128><<<(unsigned)(Nt1 / 128), 256,
      (128 * 128 + 128 * 128) * 2, stream>>>(bufB, wbf, bufA, Nt1);
  {
    int nblk = (int)((Nt1 + ROWS - 1) / ROWS);
    stats_partial_kernel<<<nblk, 256, 0, stream>>>(bufA, Nt1, 128, ROWS, part);
    stats_final_kernel<<<1, 512, 0, stream>>>(part, nblk, 128, stats);
    bnmax_kernel<<<(unsigned)(Bn * S1), 128, 0, stream>>>(
        bufA, stats, s1gb, s1bb, 1.0f / (float)Nt1, 128, S1, 0, f1pt, nullptr);
  }

  // ---- SG block 2: FPS / kNN / group on the 512 sampled points ----
  fps_kernel<<<Bn, 256, (size_t)4 * S1 * 4, stream>>>(xyz1, S1, S2, fps2, xyz2);
  knn_kernel<<<dim3(1, Bn), 256, (size_t)3 * S1 * 4, stream>>>(
      xyz1, xyz2, S1, S2, knn2);
  group_kernel<<<(unsigned)Nt2, 256, 0, stream>>>(f1pt, fps2, knn2, bufB, 128, S2, S1);

  // ---- SG2 conv a (256x256) + BN/ReLU ----
  castw_kernel<<<(256 * 256 + 255) / 256, 256, 0, stream>>>(s2wa, wbf, 256 * 256);
  gemm_wmma_kernel<256, 256><<<(unsigned)(Nt2 / 128), 256,
      (128 * 256 + 256 * 256) * 2, stream>>>(bufB, wbf, bufA, Nt2);
  bn_stage(bufA, Nt2, 256, s2ga, s2ba);

  // ---- SG2 conv b (256x256) + BN/ReLU/maxpool -> d_out [B][256][S2] ----
  castw_kernel<<<(256 * 256 + 255) / 256, 256, 0, stream>>>(s2wb, wbf, 256 * 256);
  gemm_wmma_kernel<256, 256><<<(unsigned)(Nt2 / 128), 256,
      (128 * 256 + 256 * 256) * 2, stream>>>(bufA, wbf, bufB, Nt2);
  {
    int nblk = (int)((Nt2 + ROWS - 1) / ROWS);
    stats_partial_kernel<<<nblk, 256, 0, stream>>>(bufB, Nt2, 256, ROWS, part);
    stats_final_kernel<<<1, 512, 0, stream>>>(part, nblk, 256, stats);
    bnmax_kernel<<<(unsigned)(Bn * S2), 256, 0, stream>>>(
        bufB, stats, s2gb, s2bb, 1.0f / (float)Nt2, 256, S2, 1, nullptr, out);
  }
}
